// GAT_SAGPool_45896020525321
// MI455X (gfx1250) — matrix operand order repaired and checked by reading, not compiled, not run on hardware
//
#include <hip/hip_runtime.h>

#define DIN 1024
#define HID 512
#define NN  30000
#define EE  150000

typedef __attribute__((ext_vector_type(16))) _Float16 v16h;
typedef __attribute__((ext_vector_type(8)))  _Float16 v8h;
typedef __attribute__((ext_vector_type(8)))  float    v8f;

// ---------------- utility kernels ----------------

__global__ void fill_f32(float* p, float v, int n) {
    int i = blockIdx.x * blockDim.x + threadIdx.x;
    if (i < n) p[i] = v;
}

__global__ void copy_edges(const int* __restrict__ ei, int* __restrict__ src,
                           int* __restrict__ dst, int* __restrict__ em) {
    int i = blockIdx.x * blockDim.x + threadIdx.x;
    if (i < EE) { src[i] = ei[i]; dst[i] = ei[EE + i]; em[i] = 1; }
}

__global__ void cvt_f16(const float* __restrict__ x, _Float16* __restrict__ y, int n) {
    int i = blockIdx.x * blockDim.x + threadIdx.x;
    if (i < n) y[i] = (_Float16)x[i];
}

// Wt[h][d] = W[d][h] transposed + converted to f16 (h == HID)
__global__ void cvt_w_t(const float* __restrict__ W, _Float16* __restrict__ Wt, int d) {
    int i = blockIdx.x * blockDim.x + threadIdx.x;
    if (i < d * HID) {
        int o = i / d;      // output column of W  (row of Wt)
        int r = i % d;      // input row of W
        Wt[i] = (_Float16)W[(size_t)r * HID + o];
    }
}

__device__ inline void atomicMaxF(float* addr, float v) {
    if (v >= 0.0f) atomicMax((int*)addr, __float_as_int(v));
    else           atomicMin((unsigned int*)addr, __float_as_uint(v));
}

// ---------------- WMMA GEMM: C[n x HID] = A16[n x K] * Bt16[HID x K]^T ----------------
// Block = 8 waves -> 128(M) x 64(N) tile. B tile (64 rows x 32 K halfs = 4KB) is staged
// in LDS with double-buffered global_load_async_to_lds_b128 (ASYNCcnt), each wave then
// does 4x v_wmma_f32_16x16x32_f16 per K-step, reusing one A fragment (ds-fed B).

__global__ void __launch_bounds__(256)
gemm_wmma_f16(const _Float16* __restrict__ A, const _Float16* __restrict__ Bt,
              float* __restrict__ C, int n, int K) {
    __shared__ __align__(32) _Float16 ldsB[2][64 * 32];   // 2 x 4KB

    const int tid  = threadIdx.x;
    const int lane = tid & 31;
    const int wave = tid >> 5;
    const int cb = blockIdx.x & 7;            // 8 column blocks of 64
    const int rb = blockIdx.x >> 3;           // row blocks of 128
    const int n0 = cb * 64;
    const int m0 = rb * 128 + wave * 16;
    const int half = lane >> 4;               // 0: lanes 0-15, 1: lanes 16-31
    const int l15  = lane & 15;

    // clamp A row: rows >= n compute garbage but are never stored
    int arow = m0 + l15;
    if (arow > n - 1) arow = n - 1;
    const _Float16* ap = A + (size_t)arow * K + half * 16;

    // async staging: this thread copies 16B: row srow (of 64), halfs [skoff, skoff+8)
    const int srow  = tid >> 2;
    const int skoff = (tid & 3) * 8;
    const _Float16* bsrc = Bt + (size_t)(n0 + srow) * K + skoff;
    const unsigned ldsBase =
        (unsigned)(size_t)&ldsB[0][0] + (unsigned)(srow * 32 + skoff) * 2u;

    v8f acc0 = {}, acc1 = {}, acc2 = {}, acc3 = {};
    const int nk = K >> 5;

    // prologue: stage k-block 0 into buffer 0
    {
        unsigned dst = ldsBase;
        unsigned long long g = (unsigned long long)(size_t)bsrc;
        asm volatile("global_load_async_to_lds_b128 %0, %1, off"
                     :: "v"(dst), "v"(g) : "memory");
    }

    for (int i = 0; i < nk; ++i) {
        const int kb = i << 5;
        if (i + 1 < nk) {
            unsigned dst = ldsBase + (unsigned)(((i + 1) & 1) * 4096);
            unsigned long long g = (unsigned long long)(size_t)(bsrc + kb + 32);
            asm volatile("global_load_async_to_lds_b128 %0, %1, off"
                         :: "v"(dst), "v"(g) : "memory");
            asm volatile("s_wait_asynccnt 0x1" ::: "memory");
            __builtin_prefetch(ap + kb + 32, 0, 1);
        } else {
            asm volatile("s_wait_asynccnt 0x0" ::: "memory");
        }
        __syncthreads();   // current buffer visible to all waves

        // A fragment (unconditional, clamped row)
        v8h alo = *(const v8h*)(ap + kb);        // K = base .. base+7
        v8h ahi = *(const v8h*)(ap + kb + 8);    // K = base+8 .. base+15
        v16h a;
#pragma unroll
        for (int j = 0; j < 8; ++j) { a[j] = alo[j]; a[8 + j] = ahi[j]; }

        const _Float16* lb = &ldsB[i & 1][0];
        // b fragment t: column n0+16t+l15 -> LDS row (16t+l15), halfs [half*16, +16)
        v16h b0 = *(const v16h*)(lb + (0 * 16 + l15) * 32 + half * 16);
        v16h b1 = *(const v16h*)(lb + (1 * 16 + l15) * 32 + half * 16);
        v16h b2 = *(const v16h*)(lb + (2 * 16 + l15) * 32 + half * 16);
        v16h b3 = *(const v16h*)(lb + (3 * 16 + l15) * 32 + half * 16);

        acc0 = __builtin_amdgcn_wmma_f32_16x16x32_f16(false, a, false, b0, (short)0, acc0, false, false);
        acc1 = __builtin_amdgcn_wmma_f32_16x16x32_f16(false, a, false, b1, (short)0, acc1, false, false);
        acc2 = __builtin_amdgcn_wmma_f32_16x16x32_f16(false, a, false, b2, (short)0, acc2, false, false);
        acc3 = __builtin_amdgcn_wmma_f32_16x16x32_f16(false, a, false, b3, (short)0, acc3, false, false);

        __syncthreads();   // all waves done with this buffer before it is re-staged
    }

#pragma unroll
    for (int r = 0; r < 8; ++r) {
        int m = m0 + r + half * 8;
        if (m < n) {
            float* crow = C + (size_t)m * HID + n0 + l15;
            crow[0]  = acc0[r];
            crow[16] = acc1[r];
            crow[32] = acc2[r];
            crow[48] = acc3[r];
        }
    }
}

// ---------------- GATv2 edge kernels ----------------

__global__ void __launch_bounds__(256)
edge_e_max(const float* __restrict__ xl, const float* __restrict__ xr,
           const float* __restrict__ att,
           const int* __restrict__ src, const int* __restrict__ dst,
           const int* __restrict__ em,
           float* __restrict__ e, float* __restrict__ emax, int nE, int n) {
    int lane = threadIdx.x & 31;
    int w = (blockIdx.x * blockDim.x + threadIdx.x) >> 5;
    int tot = nE + n;
    if (w >= tot) return;
    int s, d, m;
    if (w < nE) { s = src[w]; d = dst[w]; m = em[w]; }
    else        { s = d = w - nE; m = 1; }
    float v = 0.0f;
    if (m) {
        const float* pl = xl + (size_t)s * HID;
        const float* pr = xr + (size_t)d * HID;
        for (int j = lane; j < HID; j += 32) {
            float t = pl[j] + pr[j];
            t = t > 0.0f ? t : 0.2f * t;        // leaky_relu, slope 0.2
            v += t * att[j];
        }
    }
    for (int off = 16; off; off >>= 1) v += __shfl_xor(v, off, 32);
    if (lane == 0) {
        if (m) { e[w] = v; atomicMaxF(&emax[d], v); }
        else   { e[w] = -__builtin_inff(); }
    }
}

__global__ void edge_z(const float* __restrict__ e, const int* __restrict__ dst,
                       const float* __restrict__ emax, float* __restrict__ z,
                       float* __restrict__ denom, int nE, int n) {
    int i = blockIdx.x * blockDim.x + threadIdx.x;
    int tot = nE + n;
    if (i >= tot) return;
    int d = (i < nE) ? dst[i] : i - nE;
    float ev = e[i];
    float zz = (ev == -__builtin_inff()) ? 0.0f : expf(ev - emax[d]);
    z[i] = zz;
    if (zz != 0.0f) atomicAdd(&denom[d], zz);
}

__global__ void __launch_bounds__(256)
edge_agg(const float* __restrict__ xl, const int* __restrict__ src,
         const int* __restrict__ dst, const float* __restrict__ z,
         const float* __restrict__ denom, float* __restrict__ out, int nE, int n) {
    int lane = threadIdx.x & 31;
    int w = (blockIdx.x * blockDim.x + threadIdx.x) >> 5;
    int tot = nE + n;
    if (w >= tot) return;
    float zz = z[w];
    if (zz == 0.0f) return;
    int s, d;
    if (w < nE) { s = src[w]; d = dst[w]; } else { s = d = w - nE; }
    float alpha = zz / denom[d];
    const float* pl = xl + (size_t)s * HID;
    float* po = out + (size_t)d * HID;
    for (int j = lane; j < HID; j += 32) atomicAdd(&po[j], alpha * pl[j]);
}

__global__ void bias_relu(float* __restrict__ o, const float* __restrict__ b, int n) {
    int i = blockIdx.x * blockDim.x + threadIdx.x;
    if (i < n * HID) o[i] = fmaxf(o[i] + b[i % HID], 0.0f);
}

// ---------------- SAGPool kernels ----------------

__global__ void __launch_bounds__(256)
pool_score_init(const float* __restrict__ x, const float* __restrict__ Wr,
                const float* __restrict__ Wn, const float* __restrict__ b,
                float* __restrict__ attn, float* __restrict__ xwn, int n) {
    int lane = threadIdx.x & 31;
    int w = (blockIdx.x * blockDim.x + threadIdx.x) >> 5;
    if (w >= n) return;
    const float* px = x + (size_t)w * HID;
    float r = 0.0f, nv = 0.0f;
    for (int j = lane; j < HID; j += 32) {
        float xv = px[j];
        r  += xv * Wr[j];
        nv += xv * Wn[j];
    }
    for (int off = 16; off; off >>= 1) {
        r  += __shfl_xor(r, off, 32);
        nv += __shfl_xor(nv, off, 32);
    }
    if (lane == 0) { attn[w] = r + b[0]; xwn[w] = nv; }
}

__global__ void pool_scatter(const int* __restrict__ src, const int* __restrict__ dst,
                             const int* __restrict__ em, const float* __restrict__ xwn,
                             float* __restrict__ attn, int nE) {
    int i = blockIdx.x * blockDim.x + threadIdx.x;
    if (i < nE && em[i]) atomicAdd(&attn[dst[i]], xwn[src[i]]);
}

__global__ void score_key(const float* __restrict__ attn, float* __restrict__ score,
                          unsigned* __restrict__ keys, int n) {
    int i = blockIdx.x * blockDim.x + threadIdx.x;
    if (i >= n) return;
    float sc = tanhf(attn[i]);
    score[i] = sc;
    unsigned u = __float_as_uint(sc);
    u = (u & 0x80000000u) ? ~u : (u | 0x80000000u);   // order-preserving map
    keys[i] = u;
}

// single-workgroup 4-pass radix select of k-th largest key
__global__ void __launch_bounds__(256)
radix_select_k(const unsigned* __restrict__ keys, int n, int k, unsigned* __restrict__ ctr) {
    __shared__ int hist[256];
    __shared__ unsigned sprefix;
    __shared__ int skrem;
    __shared__ int g;
    int tid = threadIdx.x;
    if (tid == 0) { sprefix = 0u; skrem = k; g = 0; }
    __syncthreads();
    for (int pass = 3; pass >= 0; --pass) {
        hist[tid] = 0;
        __syncthreads();
        int shift = pass * 8;
        unsigned hm  = (pass == 3) ? 0u : (0xFFFFFFFFu << (shift + 8));
        unsigned pfx = sprefix;
        for (int i = tid; i < n; i += 256) {
            unsigned key = keys[i];
            if ((key & hm) == pfx) atomicAdd(&hist[(key >> shift) & 255], 1);
        }
        __syncthreads();
        if (tid == 0) {
            int acc = 0, b = 255;
            for (; b > 0; --b) { acc += hist[b]; if (acc >= skrem) break; }
            if (acc < skrem) acc += hist[0];           // lands in bucket 0
            skrem -= (acc - hist[b]);
            sprefix |= ((unsigned)b) << shift;
        }
        __syncthreads();
    }
    unsigned thr = sprefix;
    int loc = 0;
    for (int i = tid; i < n; i += 256) if (keys[i] > thr) loc++;
    atomicAdd(&g, loc);
    __syncthreads();
    if (tid == 0) {
        ctr[0] = 0u;                     // output counter
        ctr[1] = 0u;                     // tie counter
        ctr[2] = (unsigned)(k - g);      // ties needed
        ctr[3] = thr;                    // threshold key
    }
}

__global__ void pool_compact(const unsigned* __restrict__ keys, const float* __restrict__ score,
                             unsigned* __restrict__ ctr, int* __restrict__ mapping,
                             int* __restrict__ perm, float* __restrict__ vals, int n) {
    int i = blockIdx.x * blockDim.x + threadIdx.x;
    if (i >= n) return;
    unsigned thr = ctr[3];
    int tieNeed = (int)ctr[2];
    unsigned key = keys[i];
    int nw = -1;
    if (key > thr) {
        nw = (int)atomicAdd(&ctr[0], 1u);
    } else if (key == thr) {
        int t = (int)atomicAdd(&ctr[1], 1u);
        if (t < tieNeed) nw = (int)atomicAdd(&ctr[0], 1u);
    }
    mapping[i] = nw;
    if (nw >= 0) { perm[nw] = i; vals[nw] = score[i]; }
}

__global__ void pool_gather(const float* __restrict__ x, const int* __restrict__ perm,
                            const float* __restrict__ vals, float* __restrict__ xn, int k) {
    int i = blockIdx.x * blockDim.x + threadIdx.x;
    if (i >= k * HID) return;
    int r = i / HID, c = i % HID;
    xn[i] = x[(size_t)perm[r] * HID + c] * vals[r];
}

__global__ void edge_remap(int* __restrict__ src, int* __restrict__ dst, int* __restrict__ em,
                           const int* __restrict__ mapping, int nE) {
    int i = blockIdx.x * blockDim.x + threadIdx.x;
    if (i >= nE) return;
    int m = em[i];
    int sn = mapping[src[i]];
    int dn = mapping[dst[i]];
    int ok = (m && sn >= 0 && dn >= 0) ? 1 : 0;
    src[i] = ok ? sn : 0;
    dst[i] = ok ? dn : 0;
    em[i]  = ok;
}

// one block per column: h[c] += max over rows, h[HID+c] += mean over rows
__global__ void __launch_bounds__(256)
readout_add(const float* __restrict__ xn, int k, float* __restrict__ h) {
    __shared__ float smax[256];
    __shared__ float ssum[256];
    int c = blockIdx.x, tid = threadIdx.x;
    float mx = -__builtin_inff(), sm = 0.0f;
    for (int r = tid; r < k; r += 256) {
        float v = xn[(size_t)r * HID + c];
        mx = fmaxf(mx, v);
        sm += v;
    }
    smax[tid] = mx; ssum[tid] = sm;
    __syncthreads();
    for (int s = 128; s; s >>= 1) {
        if (tid < s) { smax[tid] = fmaxf(smax[tid], smax[tid + s]); ssum[tid] += ssum[tid + s]; }
        __syncthreads();
    }
    if (tid == 0) { h[c] += smax[0]; h[HID + c] += ssum[0] / (float)k; }
}

// ---------------- MLP head + softmax ----------------

__global__ void __launch_bounds__(512)
mlp_head(const float* __restrict__ h,
         const float* __restrict__ W0, const float* __restrict__ b0,
         const float* __restrict__ W1, const float* __restrict__ b1,
         const float* __restrict__ W2, const float* __restrict__ b2,
         float* __restrict__ out) {
    __shared__ float s1[HID];
    __shared__ float s2[HID / 2];
    __shared__ float lg[2];
    int c = threadIdx.x;
    float acc = b0[c];
    for (int i = 0; i < 2 * HID; ++i) acc += h[i] * W0[(size_t)i * HID + c];
    s1[c] = fmaxf(acc, 0.0f);
    __syncthreads();
    if (c < HID / 2) {
        float a2 = b1[c];
        for (int i = 0; i < HID; ++i) a2 += s1[i] * W1[(size_t)i * (HID / 2) + c];
        s2[c] = fmaxf(a2, 0.0f);
    }
    __syncthreads();
    if (c < 2) {
        float a3 = b2[c];
        for (int i = 0; i < HID / 2; ++i) a3 += s2[i] * W2[(size_t)i * 2 + c];
        lg[c] = a3;
        out[c] = a3;
    }
    __syncthreads();
    if (c == 0) {
        float m = fmaxf(lg[0], lg[1]);
        float e0 = expf(lg[0] - m), e1 = expf(lg[1] - m);
        out[2] = e0 / (e0 + e1);
        out[3] = e1 / (e0 + e1);
    }
}

// ---------------- host side ----------------

extern "C" void kernel_launch(void* const* d_in, const int* in_sizes, int n_in,
                              void* d_out, int out_size, void* d_ws, size_t ws_size,
                              hipStream_t stream) {
    (void)in_sizes; (void)n_in; (void)out_size; (void)ws_size;
    static const int KS_[4] = {15000, 7500, 3750, 1875};

    const float* x0 = (const float*)d_in[0];
    const int*   ei = (const int*)d_in[1];
    const float *gWl[4], *gWr[4], *gatt[4], *gb[4], *pWr[4], *pWn[4], *pb[4];
    const float *lW[3], *lb[3];
    int idx = 3;  // skip x, edge_index, batch
    for (int l = 0; l < 4; ++l) {
        gWl[l]  = (const float*)d_in[idx++];
        gWr[l]  = (const float*)d_in[idx++];
        gatt[l] = (const float*)d_in[idx++];
        gb[l]   = (const float*)d_in[idx++];
    }
    for (int l = 0; l < 4; ++l) {
        pWr[l] = (const float*)d_in[idx++];
        pWn[l] = (const float*)d_in[idx++];
        pb[l]  = (const float*)d_in[idx++];
    }
    for (int l = 0; l < 3; ++l) {
        lW[l] = (const float*)d_in[idx++];
        lb[l] = (const float*)d_in[idx++];
    }

    // carve workspace
    size_t off = 0;
    auto carve = [&](size_t bytes) -> void* {
        off = (off + 255) & ~(size_t)255;
        void* p = (char*)d_ws + off;
        off += bytes;
        return p;
    };
    // big buffer: f16 staging (n*DIN halves) aliased with GAT output (n*HID floats) — both 61.44 MB
    void*     bufBig = carve((size_t)NN * DIN * sizeof(_Float16));
    _Float16* h16  = (_Float16*)bufBig;
    float*    XO   = (float*)bufBig;
    _Float16* w16  = (_Float16*)carve((size_t)HID * DIN * sizeof(_Float16));
    float*    xl   = (float*)carve((size_t)NN * HID * sizeof(float));
    float*    xr   = (float*)carve((size_t)NN * HID * sizeof(float));
    float*    XA   = (float*)carve((size_t)15000 * HID * sizeof(float));
    float*    XB   = (float*)carve((size_t)7500  * HID * sizeof(float));
    float*    eArr = (float*)carve((size_t)(EE + NN) * sizeof(float));
    float*    zArr = (float*)carve((size_t)(EE + NN) * sizeof(float));
    float*    emax = (float*)carve((size_t)NN * sizeof(float));
    float*    deno = (float*)carve((size_t)NN * sizeof(float));
    int*      srcE = (int*)carve((size_t)EE * sizeof(int));
    int*      dstE = (int*)carve((size_t)EE * sizeof(int));
    int*      emE  = (int*)carve((size_t)EE * sizeof(int));
    float*    attn = (float*)carve((size_t)NN * sizeof(float));
    float*    xwn  = (float*)carve((size_t)NN * sizeof(float));
    float*    scor = (float*)carve((size_t)NN * sizeof(float));
    unsigned* keys = (unsigned*)carve((size_t)NN * sizeof(unsigned));
    float*    vals = (float*)carve((size_t)15000 * sizeof(float));
    int*      perm = (int*)carve((size_t)15000 * sizeof(int));
    int*      mapg = (int*)carve((size_t)NN * sizeof(int));
    float*    hvec = (float*)carve((size_t)(2 * HID) * sizeof(float));
    unsigned* ctr  = (unsigned*)carve(16 * sizeof(unsigned));

    auto G = [](int n, int b) { return (n + b - 1) / b; };
    const float NEG_INF = -__builtin_inff();

    copy_edges<<<G(EE, 256), 256, 0, stream>>>(ei, srcE, dstE, emE);
    fill_f32<<<G(2 * HID, 256), 256, 0, stream>>>(hvec, 0.0f, 2 * HID);

    int nCur = NN;
    const float* xcur = x0;
    int dcur = DIN;
    float* poolbuf[2] = {XA, XB};

    for (int l = 0; l < 4; ++l) {
        int k = KS_[l];
        // ---- GATv2: xl = x@Wl, xr = x@Wr (f16 WMMA, f32 accumulate, async-LDS B tiles) ----
        cvt_f16<<<G(nCur * dcur, 256), 256, 0, stream>>>(xcur, h16, nCur * dcur);
        int gblocks = G(nCur, 128) * (HID / 64);
        cvt_w_t<<<G(dcur * HID, 256), 256, 0, stream>>>(gWl[l], w16, dcur);
        gemm_wmma_f16<<<gblocks, 256, 0, stream>>>(h16, w16, xl, nCur, dcur);
        cvt_w_t<<<G(dcur * HID, 256), 256, 0, stream>>>(gWr[l], w16, dcur);
        gemm_wmma_f16<<<gblocks, 256, 0, stream>>>(h16, w16, xr, nCur, dcur);
        // ---- segment softmax attention + aggregate ----
        fill_f32<<<G(nCur, 256), 256, 0, stream>>>(emax, NEG_INF, nCur);
        fill_f32<<<G(nCur, 256), 256, 0, stream>>>(deno, 0.0f, nCur);
        int totE = EE + nCur;
        edge_e_max<<<G(totE * 32, 256), 256, 0, stream>>>(xl, xr, gatt[l], srcE, dstE, emE,
                                                          eArr, emax, EE, nCur);
        edge_z<<<G(totE, 256), 256, 0, stream>>>(eArr, dstE, emax, zArr, deno, EE, nCur);
        fill_f32<<<G(nCur * HID, 256), 256, 0, stream>>>(XO, 0.0f, nCur * HID);
        edge_agg<<<G(totE * 32, 256), 256, 0, stream>>>(xl, srcE, dstE, zArr, deno, XO, EE, nCur);
        bias_relu<<<G(nCur * HID, 256), 256, 0, stream>>>(XO, gb[l], nCur);
        // ---- SAGPool ----
        pool_score_init<<<G(nCur * 32, 256), 256, 0, stream>>>(XO, pWr[l], pWn[l], pb[l],
                                                               attn, xwn, nCur);
        pool_scatter<<<G(EE, 256), 256, 0, stream>>>(srcE, dstE, emE, xwn, attn, EE);
        score_key<<<G(nCur, 256), 256, 0, stream>>>(attn, scor, keys, nCur);
        radix_select_k<<<1, 256, 0, stream>>>(keys, nCur, k, ctr);
        pool_compact<<<G(nCur, 256), 256, 0, stream>>>(keys, scor, ctr, mapg, perm, vals, nCur);
        float* xn = poolbuf[l & 1];
        pool_gather<<<G(k * HID, 256), 256, 0, stream>>>(XO, perm, vals, xn, k);
        edge_remap<<<G(EE, 256), 256, 0, stream>>>(srcE, dstE, emE, mapg, EE);
        readout_add<<<HID, 256, 0, stream>>>(xn, k, hvec);
        xcur = xn;
        nCur = k;
        dcur = HID;
    }

    mlp_head<<<1, HID, 0, stream>>>(hvec, lW[0], lb[0], lW[1], lb[1], lW[2], lb[2],
                                    (float*)d_out);
}
